// Net_26852135534602
// MI455X (gfx1250) — matrix operand order repaired and checked by reading, not compiled, not run on hardware
//
#include <hip/hip_runtime.h>
#include <hip/hip_bf16.h>
#include <math.h>

typedef __attribute__((ext_vector_type(16))) _Float16 v16h;
typedef __attribute__((ext_vector_type(8)))  float    v8f;

#define NB 128
#define NN 1024
#define NE 1048576
#define NH 64

// ---------------------------------------------------------------------------
// WMMA GEMM: C[M x 64] = A[M x 64] @ W[64 x 64], all row-major f32, f16 compute
// One wave -> one 16x16 output tile, K=64 in two 16x16x32 WMMA steps.
// ---------------------------------------------------------------------------
__global__ void gemm_n64_wmma(const float* __restrict__ A,
                              const float* __restrict__ W,
                              float* __restrict__ C, int M) {
  int wave = (int)((blockIdx.x * blockDim.x + threadIdx.x) >> 5);
  int lane = threadIdx.x & 31;
  int mtiles = M >> 4;
  int tileN = wave & 3;
  int tileM = wave >> 2;
  if (tileM >= mtiles) return;

  int row  = lane & 15;      // A-row-in-tile / B,D-column-in-tile
  int half = lane >> 4;      // lane group 0/1
  const float* arow = A + (size_t)(tileM * 16 + row) * NH;
  int coln = tileN * 16 + row;

  v8f acc = {};
  #pragma unroll
  for (int kk = 0; kk < 64; kk += 32) {
    v16h a, b;
    // A 16x32 f16 as loaded here (contiguous k): lanes0-15 hold K = kk+0..15; lanes16-31 K = kk+16..31
    int ak = kk + half * 16;
    #pragma unroll
    for (int j = 0; j < 16; ++j) {
      a[j] = (_Float16)arow[ak + j];
    }
    // B 32x16 f16 as loaded here (contiguous k, same as A): lanes0-15 hold K = kk+0..15, lanes16-31 K = kk+16..31; N = lane%16
    int bk = kk + half * 16;
    #pragma unroll
    for (int j = 0; j < 16; ++j) {
      b[j] = (_Float16)W[(size_t)(bk + j) * NH + coln];
    }
    acc = __builtin_amdgcn_wmma_f32_16x16x32_f16(false, a, false, b,
                                                 (short)0, acc, false, false);
  }
  // D layout: VGPR r -> M = r + 8*half, N = lane%16
  #pragma unroll
  for (int r = 0; r < 8; ++r) {
    C[(size_t)(tileM * 16 + r + half * 8) * NH + tileN * 16 + row] = acc[r];
  }
}

// ---------------------------------------------------------------------------
// Elementwise / edge kernels
// ---------------------------------------------------------------------------
__global__ void fill_f32(float* __restrict__ p, float v, int n) {
  int i = blockIdx.x * blockDim.x + threadIdx.x;
  if (i < n) p[i] = v;
}

__global__ void init_edges(const int* __restrict__ s_in, const int* __restrict__ d_in_,
                           int* __restrict__ esrc, int* __restrict__ edst,
                           int* __restrict__ emask) {
  int e = blockIdx.x * blockDim.x + threadIdx.x;
  if (e >= NE) return;
  esrc[e] = s_in[e]; edst[e] = d_in_[e]; emask[e] = 1;
}

__global__ void deg_edge(const int* __restrict__ edst, const int* __restrict__ emask,
                         float* __restrict__ deg) {
  int e = blockIdx.x * blockDim.x + threadIdx.x;
  if (e >= NE) return;
  if (emask[e]) atomicAdd(&deg[edst[e]], 1.0f);
}

__global__ void deg_to_dis(float* __restrict__ deg, int n) {
  int i = blockIdx.x * blockDim.x + threadIdx.x;
  if (i < n) deg[i] = rsqrtf(deg[i] + 1.0f);
}

// out[v][f] = xw[v][f] * dis[v]^2 + b[f]   (self-loop message + bias)
__global__ void agg_init(const float* __restrict__ xw, const float* __restrict__ dis,
                         const float* __restrict__ b, float* __restrict__ out, int nodes) {
  int i = blockIdx.x * blockDim.x + threadIdx.x;
  if (i >= nodes * NH) return;
  int v = i >> 6, f = i & 63;
  float d = dis[v];
  out[i] = xw[i] * d * d + b[f];
}

// scatter-add neighbor messages: out[dst][f] += xw[src][f]*dis[src]*dis[dst]
__global__ void agg_edge(const float* __restrict__ xw, const float* __restrict__ dis,
                         const int* __restrict__ esrc, const int* __restrict__ edst,
                         const int* __restrict__ emask, float* __restrict__ out) {
  int idx = blockIdx.x * blockDim.x + threadIdx.x;
  int e = idx >> 6, f = idx & 63;
  if (e >= NE) return;
  if (!emask[e]) return;
  int s = esrc[e], d = edst[e];
  float norm = dis[s] * dis[d];
  atomicAdd(&out[(size_t)d * NH + f], xw[(size_t)s * NH + f] * norm);
}

__global__ void relu_k(float* __restrict__ p, int n) {
  int i = blockIdx.x * blockDim.x + threadIdx.x;
  if (i < n) p[i] = fmaxf(p[i], 0.0f);
}

// xp[v] = dot(h[v][:], Wp[:,0])
__global__ void score_x(const float* __restrict__ h, const float* __restrict__ Wp,
                        float* __restrict__ xp, int nodes) {
  int v = blockIdx.x * blockDim.x + threadIdx.x;
  if (v >= nodes) return;
  const float* r = h + (size_t)v * NH;
  float s = 0.f;
  #pragma unroll
  for (int f = 0; f < NH; ++f) s += r[f] * Wp[f];
  xp[v] = s;
}

__global__ void score_init(const float* __restrict__ xp, const float* __restrict__ dis,
                           const float* __restrict__ bp, float* __restrict__ score, int nodes) {
  int v = blockIdx.x * blockDim.x + threadIdx.x;
  if (v >= nodes) return;
  float d = dis[v];
  score[v] = xp[v] * d * d + bp[0];
}

__global__ void score_edge(const float* __restrict__ xp, const float* __restrict__ dis,
                           const int* __restrict__ esrc, const int* __restrict__ edst,
                           const int* __restrict__ emask, float* __restrict__ score) {
  int e = blockIdx.x * blockDim.x + threadIdx.x;
  if (e >= NE) return;
  if (!emask[e]) return;
  int s = esrc[e], d = edst[e];
  atomicAdd(&score[d], xp[s] * dis[s] * dis[d]);
}

// ---------------------------------------------------------------------------
// Per-graph top-k via bitonic sort in LDS (n <= 1024). One block per graph.
// Writes perm[g*k + r] = local old index, newidx[g*n + old] = global new index.
// ---------------------------------------------------------------------------
#define SORTN 1024
__global__ void pool_topk(const float* __restrict__ score, int n, int k,
                          int* __restrict__ perm, int* __restrict__ newidx) {
  __shared__ float sv[SORTN];
  __shared__ int   si[SORTN];
  int g = blockIdx.x;
  int t = threadIdx.x, bd = blockDim.x;
  for (int i = t; i < SORTN; i += bd) {
    if (i < n) { sv[i] = score[g * n + i]; newidx[g * n + i] = -1; }
    else       { sv[i] = -3.4e38f; }
    si[i] = i;
  }
  __syncthreads();
  for (int ksz = 2; ksz <= SORTN; ksz <<= 1) {
    for (int j = ksz >> 1; j > 0; j >>= 1) {
      for (int i = t; i < SORTN; i += bd) {
        int ixj = i ^ j;
        if (ixj > i) {
          bool desc = ((i & ksz) == 0);   // descending overall
          float a = sv[i], c = sv[ixj];
          bool sw = desc ? (a < c) : (a > c);
          if (sw) {
            sv[i] = c; sv[ixj] = a;
            int tmp = si[i]; si[i] = si[ixj]; si[ixj] = tmp;
          }
        }
      }
      __syncthreads();
    }
  }
  for (int r = t; r < k; r += bd) {
    int old = si[r];
    perm[g * k + r] = old;
    newidx[g * n + old] = g * k + r;
  }
}

// hout[j][f] = hin[g*n+perm[j]][f] * tanh(score[g*n+perm[j]])
__global__ void gather_pool(const float* __restrict__ hin, const float* __restrict__ score,
                            const int* __restrict__ perm, float* __restrict__ hout,
                            int n, int k) {
  int idx = blockIdx.x * blockDim.x + threadIdx.x;
  if (idx >= NB * k * NH) return;
  int f = idx & 63, j = idx >> 6;
  int g = j / k, r = j - g * k;
  int go = g * n + perm[g * k + r];
  hout[idx] = hin[(size_t)go * NH + f] * tanhf(score[go]);
}

__global__ void remap_edges(int* __restrict__ esrc, int* __restrict__ edst,
                            int* __restrict__ emask, const int* __restrict__ newidx) {
  int e = blockIdx.x * blockDim.x + threadIdx.x;
  if (e >= NE) return;
  if (emask[e]) {
    int ns = newidx[esrc[e]], nd = newidx[edst[e]];
    if (ns >= 0 && nd >= 0) { esrc[e] = ns; edst[e] = nd; }
    else { esrc[e] = 0; edst[e] = 0; emask[e] = 0; }
  } else { esrc[e] = 0; edst[e] = 0; }
}

// zsum[g][0:64] += max over k nodes; zsum[g][64:128] += mean over k nodes
__global__ void readout_accum(const float* __restrict__ h, int k, float* __restrict__ zsum) {
  int idx = blockIdx.x * blockDim.x + threadIdx.x;
  if (idx >= NB * NH) return;
  int g = idx >> 6, f = idx & 63;
  const float* p = h + (size_t)(g * k) * NH + f;
  float mx = -3.4e38f, sm = 0.f;
  for (int r = 0; r < k; ++r) { float v = p[(size_t)r * NH]; mx = fmaxf(mx, v); sm += v; }
  zsum[g * 128 + f]      += mx;
  zsum[g * 128 + 64 + f] += sm / (float)k;
}

// ---------------------------------------------------------------------------
// MLP head
// ---------------------------------------------------------------------------
__global__ void mlp1(const float* __restrict__ z, const float* __restrict__ W,
                     const float* __restrict__ b, float* __restrict__ o) {
  int idx = blockIdx.x * blockDim.x + threadIdx.x;
  if (idx >= NB * 64) return;
  int g = idx >> 6, oo = idx & 63;
  float s = b[oo];
  #pragma unroll 4
  for (int i = 0; i < 128; ++i) s += z[g * 128 + i] * W[i * 64 + oo];
  o[idx] = fmaxf(s, 0.f);
}

__global__ void mlp2(const float* __restrict__ z, const float* __restrict__ W,
                     const float* __restrict__ b, float* __restrict__ o) {
  int idx = blockIdx.x * blockDim.x + threadIdx.x;
  if (idx >= NB * 32) return;
  int g = idx >> 5, oo = idx & 31;
  float s = b[oo];
  #pragma unroll 4
  for (int i = 0; i < 64; ++i) s += z[g * 64 + i] * W[i * 32 + oo];
  o[idx] = fmaxf(s, 0.f);
}

__global__ void mlp3_lsm(const float* __restrict__ z, const float* __restrict__ W,
                         const float* __restrict__ b, float* __restrict__ out) {
  int g = blockIdx.x * blockDim.x + threadIdx.x;
  if (g >= NB) return;
  float logit[10];
  float mx = -3.4e38f;
  for (int o = 0; o < 10; ++o) {
    float s = b[o];
    for (int i = 0; i < 32; ++i) s += z[g * 32 + i] * W[i * 10 + o];
    logit[o] = s; mx = fmaxf(mx, s);
  }
  float se = 0.f;
  for (int o = 0; o < 10; ++o) se += expf(logit[o] - mx);
  float lse = logf(se) + mx;
  for (int o = 0; o < 10; ++o) out[g * 10 + o] = logit[o] - lse;
}

// ---------------------------------------------------------------------------
extern "C" void kernel_launch(void* const* d_in, const int* in_sizes, int n_in,
                              void* d_out, int out_size, void* d_ws, size_t ws_size,
                              hipStream_t stream) {
  (void)in_sizes; (void)n_in; (void)out_size; (void)ws_size;
  const float* x      = (const float*)d_in[0];
  const int*   src_in = (const int*)d_in[1];
  const int*   dst_in = (const int*)d_in[2];
  const float* W1 = (const float*)d_in[3];  const float* b1  = (const float*)d_in[4];
  const float* Wp1= (const float*)d_in[5];  const float* bp1 = (const float*)d_in[6];
  const float* W2 = (const float*)d_in[7];  const float* b2  = (const float*)d_in[8];
  const float* Wp2= (const float*)d_in[9];  const float* bp2 = (const float*)d_in[10];
  const float* W3 = (const float*)d_in[11]; const float* b3  = (const float*)d_in[12];
  const float* Wp3= (const float*)d_in[13]; const float* bp3 = (const float*)d_in[14];
  const float* L1w= (const float*)d_in[15]; const float* L1b = (const float*)d_in[16];
  const float* L2w= (const float*)d_in[17]; const float* L2b = (const float*)d_in[18];
  const float* L3w= (const float*)d_in[19]; const float* L3b = (const float*)d_in[20];

  char* w = (char*)d_ws;
  auto alloc = [&](size_t bytes) -> void* {
    void* p = (void*)w; w += (bytes + 255) & ~(size_t)255; return p;
  };
  const size_t FEAT = (size_t)NB * NN * NH * sizeof(float); // 32 MB
  float* hA    = (float*)alloc(FEAT);                 // pooled features (stage output)
  float* xw    = (float*)alloc(FEAT);                 // GEMM output
  float* hC    = (float*)alloc(FEAT);                 // conv output (post relu)
  float* dis   = (float*)alloc((size_t)NB * NN * 4);  // deg -> rsqrt(deg+1)
  float* xp    = (float*)alloc((size_t)NB * NN * 4);
  float* score = (float*)alloc((size_t)NB * NN * 4);
  int*   newidx= (int*)  alloc((size_t)NB * NN * 4);
  int*   perm  = (int*)  alloc((size_t)NB * NN * 4);
  int*   esrc  = (int*)  alloc((size_t)NE * 4);
  int*   edst  = (int*)  alloc((size_t)NE * 4);
  int*   emask = (int*)  alloc((size_t)NE * 4);
  float* zsum  = (float*)alloc((size_t)NB * 128 * 4);
  float* z1    = (float*)alloc((size_t)NB * 64 * 4);
  float* z2    = (float*)alloc((size_t)NB * 32 * 4);

  const int T = 256;
  fill_f32<<<(NB * 128 + T - 1) / T, T, 0, stream>>>(zsum, 0.f, NB * 128);
  init_edges<<<(NE + T - 1) / T, T, 0, stream>>>(src_in, dst_in, esrc, edst, emask);

  auto stage = [&](const float* hin, int n, int k, const float* Wm, const float* bm,
                   const float* Wp, const float* bp) {
    int nodes = NB * n;
    int felems = nodes * NH;
    // degree + symmetric normalization
    fill_f32<<<(nodes + T - 1) / T, T, 0, stream>>>(dis, 0.f, nodes);
    deg_edge<<<(NE + T - 1) / T, T, 0, stream>>>(edst, emask, dis);
    deg_to_dis<<<(nodes + T - 1) / T, T, 0, stream>>>(dis, nodes);
    // xw = h @ W  (WMMA)
    int tiles = (nodes / 16) * 4;
    gemm_n64_wmma<<<(tiles + 3) / 4, 128, 0, stream>>>(hin, Wm, xw, nodes);
    // aggregate: self + neighbors + bias, then relu
    agg_init<<<(felems + T - 1) / T, T, 0, stream>>>(xw, dis, bm, hC, nodes);
    agg_edge<<<(NE * NH + T - 1) / T, T, 0, stream>>>(xw, dis, esrc, edst, emask, hC);
    relu_k<<<(felems + T - 1) / T, T, 0, stream>>>(hC, felems);
    // pooling score = GCNConv(h, Wp, bp)
    score_x<<<(nodes + T - 1) / T, T, 0, stream>>>(hC, Wp, xp, nodes);
    score_init<<<(nodes + T - 1) / T, T, 0, stream>>>(xp, dis, bp, score, nodes);
    score_edge<<<(NE + T - 1) / T, T, 0, stream>>>(xp, dis, esrc, edst, emask, score);
    // top-k select, gate+gather, remap edges, readout accumulate
    pool_topk<<<NB, 256, 0, stream>>>(score, n, k, perm, newidx);
    int gelems = NB * k * NH;
    gather_pool<<<(gelems + T - 1) / T, T, 0, stream>>>(hC, score, perm, hA, n, k);
    remap_edges<<<(NE + T - 1) / T, T, 0, stream>>>(esrc, edst, emask, newidx);
    readout_accum<<<(NB * NH + T - 1) / T, T, 0, stream>>>(hA, k, zsum);
  };

  const int K1 = 820, K2 = 656, K3 = 525;   // ceil(0.8 * n) chain
  stage(x,  NN, K1, W1, b1, Wp1, bp1);
  stage(hA, K1, K2, W2, b2, Wp2, bp2);
  stage(hA, K2, K3, W3, b3, Wp3, bp3);

  mlp1<<<(NB * 64 + T - 1) / T, T, 0, stream>>>(zsum, L1w, L1b, z1);
  mlp2<<<(NB * 32 + T - 1) / T, T, 0, stream>>>(z1, L2w, L2b, z2);
  mlp3_lsm<<<1, NB, 0, stream>>>(z2, L3w, L3b, (float*)d_out);
}
